// VQ_VAE_87557203296354
// MI455X (gfx1250) — compile-verified, matrix operand order checked by reading, not run on hardware
//
#include <hip/hip_runtime.h>
#include <hip/hip_bf16.h>

typedef __attribute__((ext_vector_type(16))) _Float16 v16h;
typedef __attribute__((ext_vector_type(8)))  _Float16 v8h;
typedef __attribute__((ext_vector_type(4)))  _Float16 v4h;
typedef __attribute__((ext_vector_type(8)))  float    v8f;

#define BM 128
#define BN 128
#define BK 64
#define LDH 72   // f16 row stride in LDS: 144B rows, 16B aligned, conflict-free banks

enum { EPI_RELU_H = 0, EPI_NONE_F = 1, EPI_SIG_F = 2 };

// Async global->LDS 16-byte copy (CDNA5). LDS aperture: low 32 bits of a flat
// shared pointer are the DS byte offset.
__device__ __forceinline__ void async_cp_b128(_Float16* lds, const _Float16* g) {
  unsigned loff = (unsigned)(unsigned long long)lds;
  asm volatile("global_load_async_to_lds_b128 %0, %1, off"
               :: "v"(loff), "v"(g)
               : "memory");
}
__device__ __forceinline__ void zero_b128(_Float16* lds) {
  v8h z = {};
  *(v8h*)lds = z;
}

// C[M,N] = act(A[M,K] @ W[N,K]^T + bias[N]); A, W f16; K % 8 == 0, K >= 64.
// Block 128x128, 8 waves; wave tile 32x64 (2x4 WMMA accumulators).
template <int EPI>
__global__ __launch_bounds__(256) void gemm_wmma_f16(
    const _Float16* __restrict__ A, int lda,
    const _Float16* __restrict__ W,
    const float* __restrict__ bias,
    void* __restrict__ Cout, int ldc, int N, int K) {
  __shared__ __align__(16) _Float16 As[2][BM * LDH];
  __shared__ __align__(16) _Float16 Bs[2][BN * LDH];

  const int t    = threadIdx.x;
  const int m0   = blockIdx.y * BM;
  const int n0   = blockIdx.x * BN;
  const int lane = t & 31;
  const int wv   = t >> 5;           // 8 waves
  const int wm   = (wv & 3) * 32;    // 4 waves over M
  const int wn   = (wv >> 2) * 64;   // 2 waves over N
  const int hl   = lane >> 4;
  const int l16  = lane & 15;

  // staging: 2 threads/row, 4 x 16B chunks each (both tiles are 128 x 64 f16)
  const int rl  = t >> 1;            // 0..127
  const int kf  = (t & 1) * 32;      // 0 / 32
  const size_t aG = (size_t)(m0 + rl) * (size_t)lda;
  const int    bn = (n0 + rl) < N ? (n0 + rl) : (N - 1);  // clamp; cols masked in epilogue
  const size_t bG = (size_t)bn * (size_t)K;

  const int nfull   = K / BK;
  const int nstages = nfull + ((K % BK) ? 1 : 0);

  auto issue_full = [&](int s, int buf) {
    const int kk = s * BK + kf;
    _Float16* ad = &As[buf][rl * LDH + kf];
    _Float16* bd = &Bs[buf][rl * LDH + kf];
#pragma unroll
    for (int c = 0; c < 4; ++c) async_cp_b128(ad + c * 8, &A[aG + kk + c * 8]);
#pragma unroll
    for (int c = 0; c < 4; ++c) async_cp_b128(bd + c * 8, &W[bG + kk + c * 8]);
  };
  auto issue_tail = [&](int buf) {
    const int kk = nfull * BK + kf;
    _Float16* ad = &As[buf][rl * LDH + kf];
    _Float16* bd = &Bs[buf][rl * LDH + kf];
#pragma unroll
    for (int c = 0; c < 4; ++c) {
      if (kk + c * 8 < K) async_cp_b128(ad + c * 8, &A[aG + kk + c * 8]);
      else                zero_b128(ad + c * 8);
    }
#pragma unroll
    for (int c = 0; c < 4; ++c) {
      if (kk + c * 8 < K) async_cp_b128(bd + c * 8, &W[bG + kk + c * 8]);
      else                zero_b128(bd + c * 8);
    }
  };

  v8f acc[2][4] = {};
  int cur = 0;
  issue_full(0, 0);

  for (int s = 0; s < nstages; ++s) {
    if (s + 1 < nfull) {
      issue_full(s + 1, cur ^ 1);
      // in-order async retire: <=8 outstanding => stage s fully resident
      asm volatile("s_wait_asynccnt 0x8" ::: "memory");
    } else if (s + 1 < nstages) {
      issue_tail(cur ^ 1);            // tail issues a divergent count -> drain
      asm volatile("s_wait_asynccnt 0x0" ::: "memory");
    } else {
      asm volatile("s_wait_asynccnt 0x0" ::: "memory");
    }
    __syncthreads();

    union { v16h v; v8h h[2]; } u;
#pragma unroll
    for (int ks = 0; ks < BK; ks += 32) {
      v16h af[2], bf[4];
#pragma unroll
      for (int i = 0; i < 2; ++i) {
        const _Float16* p = &As[cur][(wm + i * 16 + l16) * LDH + ks];
        u.h[0] = *(const v8h*)(p + hl * 8);        // K 0..7 / 8..15
        u.h[1] = *(const v8h*)(p + 16 + hl * 8);   // K 16..23 / 24..31
        af[i] = u.v;
      }
#pragma unroll
      for (int j = 0; j < 4; ++j) {
        const _Float16* p = &Bs[cur][(wn + j * 16 + l16) * LDH + ks];
        u.h[0] = *(const v8h*)(p + hl * 8);
        u.h[1] = *(const v8h*)(p + 16 + hl * 8);
        bf[j] = u.v;
      }
#pragma unroll
      for (int i = 0; i < 2; ++i)
#pragma unroll
        for (int j = 0; j < 4; ++j)
          acc[i][j] = __builtin_amdgcn_wmma_f32_16x16x32_f16(
              false, af[i], false, bf[j], (short)0, acc[i][j], false, false);
    }
    __syncthreads();   // all waves done with 'cur' before it is re-issued
    cur ^= 1;
  }

  // fused epilogue: C layout m = r + 8*hl, n = l16
#pragma unroll
  for (int j = 0; j < 4; ++j) {
    const int n = n0 + wn + j * 16 + l16;
    if (n < N) {
      const float bv = bias[n];
#pragma unroll
      for (int i = 0; i < 2; ++i) {
        const int mbase = m0 + wm + i * 16 + hl * 8;
#pragma unroll
        for (int r = 0; r < 8; ++r) {
          float v = acc[i][j][r] + bv;
          const size_t off = (size_t)(mbase + r) * (size_t)ldc + (size_t)n;
          if constexpr (EPI == EPI_RELU_H) {
            ((_Float16*)Cout)[off] = (_Float16)fmaxf(v, 0.f);
          } else if constexpr (EPI == EPI_NONE_F) {
            ((float*)Cout)[off] = v;
          } else {
            ((float*)Cout)[off] = 1.f / (1.f + __expf(-v));
          }
        }
      }
    }
  }
}

// Vectorized fp32 -> f16 conversion, 4 elements/thread (n % 4 == 0).
__global__ __launch_bounds__(256) void cvt_f32_f16(const float* __restrict__ s,
                                                   _Float16* __restrict__ d,
                                                   long n4) {
  const long i = (long)blockIdx.x * 256 + threadIdx.x;
  if (i < n4) {
    const float4 v = ((const float4*)s)[i];
    v4h o;
    o[0] = (_Float16)v.x;
    o[1] = (_Float16)v.y;
    o[2] = (_Float16)v.z;
    o[3] = (_Float16)v.w;
    ((v4h*)d)[i] = o;
  }
}

// VQ nearest-codebook lookup. z_e laid out (B, 10, 20) row-major.
__global__ __launch_bounds__(256) void vq_quantize(const float* __restrict__ z_e,
                                                   const float* __restrict__ emb,  // (10,16)
                                                   float* __restrict__ z_q,
                                                   _Float16* __restrict__ z_q_h,
                                                   int total) {
  __shared__ float se[160];
  __shared__ float ssq[16];
  const int t = threadIdx.x;
  if (t < 160) se[t] = emb[t];
  __syncthreads();
  if (t < 16) {
    float s = 0.f;
#pragma unroll
    for (int d = 0; d < 10; ++d) s += se[d * 16 + t] * se[d * 16 + t];
    ssq[t] = s;
  }
  __syncthreads();

  const int g = blockIdx.x * 256 + t;
  if (g >= total) return;
  const int b = g / 20, p = g % 20;
  const size_t base = (size_t)b * 200 + (size_t)p;

  float f[10];
#pragma unroll
  for (int d = 0; d < 10; ++d) f[d] = z_e[base + d * 20];

  int best = 0;
  float bd = 3.4e38f;
#pragma unroll
  for (int j = 0; j < 16; ++j) {
    float dist = ssq[j];
#pragma unroll
    for (int d = 0; d < 10; ++d) dist -= 2.f * f[d] * se[d * 16 + j];
    if (dist < bd) { bd = dist; best = j; }
  }
#pragma unroll
  for (int d = 0; d < 10; ++d) {
    const float q = se[d * 16 + best];
    z_q[base + d * 20]   = q;
    z_q_h[base + d * 20] = (_Float16)q;
  }
}

extern "C" void kernel_launch(void* const* d_in, const int* in_sizes, int n_in,
                              void* d_out, int out_size, void* d_ws, size_t ws_size,
                              hipStream_t stream) {
  const float* x   = (const float*)d_in[0];
  const float* w1  = (const float*)d_in[1];
  const float* b1  = (const float*)d_in[2];
  const float* w2  = (const float*)d_in[3];
  const float* b2  = (const float*)d_in[4];
  const float* w3  = (const float*)d_in[5];
  const float* b3  = (const float*)d_in[6];
  const float* w4  = (const float*)d_in[7];
  const float* b4  = (const float*)d_in[8];
  const float* emb = (const float*)d_in[9];

  const int B = in_sizes[0] / 784;

  float* out   = (float*)d_out;
  float* recon = out;                          // B*784
  float* z_e   = out + (size_t)B * 784;        // B*200
  float* z_q   = z_e + (size_t)B * 200;        // B*200

  // workspace layout (all f16)
  char* ws = (char*)d_ws;
  size_t o = 0;
  _Float16* xh  = (_Float16*)(ws + o); o += (size_t)B * 784 * 2;
  _Float16* h1  = (_Float16*)(ws + o); o += (size_t)B * 400 * 2;
  _Float16* zqh = (_Float16*)(ws + o); o += (size_t)B * 200 * 2;
  _Float16* h3  = (_Float16*)(ws + o); o += (size_t)B * 400 * 2;
  _Float16* w1h = (_Float16*)(ws + o); o += (size_t)400 * 784 * 2;
  _Float16* w2h = (_Float16*)(ws + o); o += (size_t)200 * 400 * 2;
  _Float16* w3h = (_Float16*)(ws + o); o += (size_t)400 * 200 * 2;
  _Float16* w4h = (_Float16*)(ws + o);

  const dim3 blk(256);
  const int mg = B / BM;  // 512

  // fp32 -> f16 conversions (x and weights); all counts % 4 == 0
  {
    const long nx = (long)B * 784 / 4;
    cvt_f32_f16<<<(unsigned)((nx + 255) / 256), blk, 0, stream>>>(x, xh, nx);
    const long n1 = 400L * 784 / 4;
    cvt_f32_f16<<<(unsigned)((n1 + 255) / 256), blk, 0, stream>>>(w1, w1h, n1);
    const long n2 = 200L * 400 / 4;
    cvt_f32_f16<<<(unsigned)((n2 + 255) / 256), blk, 0, stream>>>(w2, w2h, n2);
    const long n3 = 400L * 200 / 4;
    cvt_f32_f16<<<(unsigned)((n3 + 255) / 256), blk, 0, stream>>>(w3, w3h, n3);
    const long n4 = 784L * 400 / 4;
    cvt_f32_f16<<<(unsigned)((n4 + 255) / 256), blk, 0, stream>>>(w4, w4h, n4);
  }

  // L1: h1 = relu(x @ w1^T + b1)        (K=784, N=400)
  gemm_wmma_f16<EPI_RELU_H><<<dim3((400 + BN - 1) / BN, mg), blk, 0, stream>>>(
      xh, 784, w1h, b1, (void*)h1, 400, 400, 784);
  // L2: z_e = h1 @ w2^T + b2            (K=400, N=200) -> f32 output
  gemm_wmma_f16<EPI_NONE_F><<<dim3((200 + BN - 1) / BN, mg), blk, 0, stream>>>(
      h1, 400, w2h, b2, (void*)z_e, 200, 200, 400);
  // VQ: nearest codebook per (b, pos)
  {
    const int total = B * 20;
    vq_quantize<<<(total + 255) / 256, blk, 0, stream>>>(z_e, emb, z_q, zqh, total);
  }
  // L3: h3 = relu(z_q @ w3^T + b3)      (K=200, N=400)
  gemm_wmma_f16<EPI_RELU_H><<<dim3((400 + BN - 1) / BN, mg), blk, 0, stream>>>(
      zqh, 200, w3h, b3, (void*)h3, 400, 400, 200);
  // L4: recon = sigmoid(h3 @ w4^T + b4) (K=400, N=784) -> f32 output
  gemm_wmma_f16<EPI_SIG_F><<<dim3((784 + BN - 1) / BN, mg), blk, 0, stream>>>(
      h3, 400, w4h, b4, (void*)recon, 784, 784, 400);
}
